// MultiHeadAttention_41712722379083
// MI455X (gfx1250) — compile-verified, matrix operand order checked by reading, not run on hardware
//
#include <hip/hip_runtime.h>
#include <hip/hip_bf16.h>
#include <stdint.h>

// ---------------------------------------------------------------------------
// MHA forward for MI455X (gfx1250, wave32).
//   B=4, S=2048, D=512, H=8, HD=64
// Pipeline:
//   0) cvt_kernel    : one-time f32 -> bf16 conversion of inputs + weights
//   1) proj_kernel   : Q/K/V = X @ W^T + b   (pure bf16 WMMA, V transposed)
//   2) attn_kernel   : logits -> softmax (2-pass flash stats) -> attn out + ctx
//   3) outproj_kernel: out = ctx @ Wo^T + bo (f32 out)
// All matmuls use v_wmma_f32_16x16x32_bf16 (f32 accumulate).
// ---------------------------------------------------------------------------

#define DEV __device__ __forceinline__

typedef __attribute__((ext_vector_type(16))) __bf16 v16bf;
typedef __attribute__((ext_vector_type(8)))  float  v8f;

static constexpr int kB  = 4;
static constexpr int kS  = 2048;
static constexpr int kD  = 512;
static constexpr int kH  = 8;
static constexpr int kHD = 64;

union FragAB {
    v16bf v;
    unsigned short u[16];
    uint4 q[2];
};
union FragC {
    v8f   v;
    float f[8];
};

// float -> bf16 round-to-nearest-even (bit manipulation)
DEV unsigned short f2bf(float f) {
    union { float f; uint32_t u; } x;
    x.f = f;
    uint32_t r = x.u + 0x7FFFu + ((x.u >> 16) & 1u);
    return (unsigned short)(r >> 16);
}

DEV v8f wmma_bf16(v16bf a, v16bf b, v8f c) {
    // (neg_a, A, neg_b, B, c_mod, C, reuse_a, reuse_b)
    return __builtin_amdgcn_wmma_f32_16x16x32_bf16(false, a, false, b,
                                                   (short)0, c, false, false);
}

// Load a 16x32 (MxK) A-fragment / 32x16 (KxN) B-fragment from a row-major
// bf16 matrix. ISA layout (16-bit operands, wave32):
//   lane l : row/col = l&15 ; hi = (l>=16)?8:0
//   halves j=0..7  -> k = k0 + hi + j        (contiguous 8 -> one b128)
//   halves j=8..15 -> k = k0 + 16 + hi + j-8 (contiguous 8 -> one b128)
DEV v16bf load_frag_bf16(const unsigned short* base, size_t ld, int row,
                         int k0, int lane) {
    int hi = (lane >> 4) << 3;
    const unsigned short* p = base + (size_t)row * ld + k0 + hi;
    FragAB fr;
    fr.q[0] = *(const uint4*)(p);
    fr.q[1] = *(const uint4*)(p + 16);
    return fr.v;
}

// ---------------------------------------------------------------------------
// Kernel 0: elementwise f32 -> bf16 (8 elements / thread, b128 in, b128 out
// per thread: 2 loads, 1 store). n must be a multiple of 8 (all our sizes are).
// ---------------------------------------------------------------------------
__global__ __launch_bounds__(256) void mha_cvt_kernel(
    const float* __restrict__ src, unsigned short* __restrict__ dst, int n8) {
    int i = blockIdx.x * blockDim.x + threadIdx.x;
    if (i >= n8) return;
    const float4* s = (const float4*)src + (size_t)i * 2;
    float4 a = s[0];
    float4 b = s[1];
    FragAB half;  // only u[0..7]/q[0] used
    half.u[0] = f2bf(a.x); half.u[1] = f2bf(a.y);
    half.u[2] = f2bf(a.z); half.u[3] = f2bf(a.w);
    half.u[4] = f2bf(b.x); half.u[5] = f2bf(b.y);
    half.u[6] = f2bf(b.z); half.u[7] = f2bf(b.w);
    *(uint4*)(dst + (size_t)i * 8) = half.q[0];
}

// ---------------------------------------------------------------------------
// Kernel 1: Q/K/V projections.  grid = (64, 8, 3), block = 256 (8 waves).
// Wave computes a 16x64 tile of C = X @ W^T + b (all operands bf16).
// z=0 -> Q (layout [B,H,S,HD] bf16), z=1 -> K (same), z=2 -> V^T [B,H,HD,S].
// ---------------------------------------------------------------------------
__global__ __launch_bounds__(256) void mha_proj_kernel(
    const unsigned short* __restrict__ Xq, const unsigned short* __restrict__ Xk,
    const unsigned short* __restrict__ Xv, const unsigned short* __restrict__ Wq,
    const unsigned short* __restrict__ Wk, const unsigned short* __restrict__ Wv,
    const float* __restrict__ bq, const float* __restrict__ bk,
    const float* __restrict__ bv, unsigned short* __restrict__ Qb,
    unsigned short* __restrict__ Kb, unsigned short* __restrict__ Vt) {
    const int lane = threadIdx.x & 31;
    const int wave = threadIdx.x >> 5;
    const int z    = blockIdx.z;
    const int m0   = (blockIdx.x * 8 + wave) * 16;   // token-row block
    const int n0   = blockIdx.y * 64;                // output-column block

    const unsigned short* X = (z == 0) ? Xq : (z == 1) ? Xk : Xv;
    const unsigned short* W = (z == 0) ? Wq : (z == 1) ? Wk : Wv;
    const float* bias = (z == 0) ? bq : (z == 1) ? bk : bv;

    FragC acc[4];
#pragma unroll
    for (int t = 0; t < 4; ++t)
#pragma unroll
        for (int i = 0; i < 8; ++i) acc[t].f[i] = 0.0f;

    const int rowA = m0 + (lane & 15);
    for (int k0 = 0; k0 < kD; k0 += 32) {
        v16bf a = load_frag_bf16(X, kD, rowA, k0, lane);
#pragma unroll
        for (int t = 0; t < 4; ++t) {
            v16bf bf = load_frag_bf16(W, kD, n0 + t * 16 + (lane & 15), k0, lane);
            acc[t].v = wmma_bf16(a, bf, acc[t].v);
        }
    }

    const int hi8 = (lane >> 4) << 3;
    const int col = lane & 15;
#pragma unroll
    for (int t = 0; t < 4; ++t) {
#pragma unroll
        for (int r = 0; r < 8; ++r) {
            int mrow = m0 + r + hi8;
            int ncol = n0 + t * 16 + col;
            unsigned short val = f2bf(acc[t].f[r] + bias[ncol]);
            int b = mrow >> 11, s = mrow & (kS - 1);
            int h = ncol >> 6, hd = ncol & (kHD - 1);
            size_t bh = (size_t)(b * kH + h);
            if (z == 0)
                Qb[(bh * kS + s) * kHD + hd] = val;
            else if (z == 1)
                Kb[(bh * kS + s) * kHD + hd] = val;
            else
                Vt[(bh * kHD + hd) * kS + s] = val;   // transposed
        }
    }
}

// ---------------------------------------------------------------------------
// Kernel 2: attention.  grid = (16, 32), block = 256 (8 waves).
// Wave owns 16 query rows of one (b,h). Two passes over keys:
//   pass 1: QK^T (WMMA) + mask -> online row max / sum (shfl reduction)
//   pass 2: recompute QK^T, write normalized attn (NT stores), accumulate
//           ctx = P @ V via WMMA (P transposed C->A layout through LDS)
// ---------------------------------------------------------------------------
__global__ __launch_bounds__(256) void mha_attn_kernel(
    const unsigned short* __restrict__ Qb, const unsigned short* __restrict__ Kb,
    const unsigned short* __restrict__ Vt, const float* __restrict__ mask,
    float* __restrict__ attn, unsigned short* __restrict__ ctx) {
    __shared__ unsigned short ptile[8][16][32];

    const int lane = threadIdx.x & 31;
    const int wave = threadIdx.x >> 5;
    const int bh   = blockIdx.y;
    const int b    = bh >> 3;
    const int h    = bh & 7;
    const int q0   = (blockIdx.x * 8 + wave) * 16;

    const unsigned short* Qh = Qb + (size_t)bh * kS * kHD;
    const unsigned short* Kh = Kb + (size_t)bh * kS * kHD;
    const unsigned short* Vh = Vt + (size_t)bh * kHD * kS;
    const float* mbase = mask + (size_t)b * kS * kS;
    float* abase = attn + (size_t)bh * kS * kS;

    const int hi8 = (lane >> 4) << 3;
    const int col = lane & 15;
    const float scale = 0.125f;  // 1/sqrt(64)

    // Q A-fragments for the whole 16-row block (HD = 64 -> two k=32 steps)
    v16bf aq0 = load_frag_bf16(Qh, kHD, q0 + col, 0, lane);
    v16bf aq1 = load_frag_bf16(Qh, kHD, q0 + col, 32, lane);

    float mstat[8], sstat[8];
#pragma unroll
    for (int r = 0; r < 8; ++r) { mstat[r] = -3.0e38f; sstat[r] = 0.0f; }

    // ---- pass 1: softmax statistics --------------------------------------
    for (int kt = 0; kt < kS; kt += 16) {
        v16bf bk0 = load_frag_bf16(Kh, kHD, kt + col, 0, lane);
        v16bf bk1 = load_frag_bf16(Kh, kHD, kt + col, 32, lane);
        FragC c;
#pragma unroll
        for (int i = 0; i < 8; ++i) c.f[i] = 0.0f;
        c.v = wmma_bf16(aq0, bk0, c.v);
        c.v = wmma_bf16(aq1, bk1, c.v);

        int key = kt + col;
#pragma unroll
        for (int r = 0; r < 8; ++r) {
            int qrow = q0 + r + hi8;
            float mk = __builtin_nontemporal_load(
                &mbase[(size_t)qrow * kS + key]);
            float logit = c.f[r] * scale - 1.0e9f * mk;
            float vmax = logit;
#pragma unroll
            for (int off = 1; off < 16; off <<= 1)
                vmax = fmaxf(vmax, __shfl_xor(vmax, off, 32));
            float mnew = fmaxf(mstat[r], vmax);
            float p = __expf(logit - mnew);
            float ps = p;
#pragma unroll
            for (int off = 1; off < 16; off <<= 1)
                ps += __shfl_xor(ps, off, 32);
            sstat[r] = sstat[r] * __expf(mstat[r] - mnew) + ps;
            mstat[r] = mnew;
        }
    }

    float rinv[8];
#pragma unroll
    for (int r = 0; r < 8; ++r) rinv[r] = 1.0f / sstat[r];

    FragC oacc[4];
#pragma unroll
    for (int t = 0; t < 4; ++t)
#pragma unroll
        for (int i = 0; i < 8; ++i) oacc[t].f[i] = 0.0f;

    // ---- pass 2: recompute logits, write attn, accumulate ctx = P @ V ----
    for (int kt = 0; kt < kS; kt += 32) {
#pragma unroll
        for (int half = 0; half < 2; ++half) {
            int kbase = kt + half * 16;
            v16bf bk0 = load_frag_bf16(Kh, kHD, kbase + col, 0, lane);
            v16bf bk1 = load_frag_bf16(Kh, kHD, kbase + col, 32, lane);
            FragC c;
#pragma unroll
            for (int i = 0; i < 8; ++i) c.f[i] = 0.0f;
            c.v = wmma_bf16(aq0, bk0, c.v);
            c.v = wmma_bf16(aq1, bk1, c.v);

            int key = kbase + col;
#pragma unroll
            for (int r = 0; r < 8; ++r) {
                int qrow = q0 + r + hi8;
                float mk = __builtin_nontemporal_load(
                    &mbase[(size_t)qrow * kS + key]);
                float logit = c.f[r] * scale - 1.0e9f * mk;
                float p = __expf(logit - mstat[r]) * rinv[r];
                __builtin_nontemporal_store(p,
                    &abase[(size_t)qrow * kS + key]);
                ptile[wave][r + hi8][half * 16 + col] = f2bf(p);
            }
        }

        // per-wave LDS transpose: C-layout P tile -> A-layout fragment
        __asm volatile("s_wait_dscnt 0x0" ::: "memory");
        const unsigned short* lp = &ptile[wave][col][0];
        FragAB pa;
        pa.q[0] = *(const uint4*)(lp + hi8);
        pa.q[1] = *(const uint4*)(lp + 16 + hi8);

#pragma unroll
        for (int t = 0; t < 4; ++t) {
            // V^T row = head-dim index, contiguous along keys
            v16bf vb = load_frag_bf16(Vh, kS, t * 16 + col, kt, lane);
            oacc[t].v = wmma_bf16(pa.v, vb, oacc[t].v);
        }
    }

    // store ctx (bf16, [B*S, D] with col = h*64 + d)
#pragma unroll
    for (int t = 0; t < 4; ++t) {
#pragma unroll
        for (int r = 0; r < 8; ++r) {
            int qrow = q0 + r + hi8;
            int dcol = h * kHD + t * 16 + col;
            ctx[((size_t)(b * kS + qrow)) * kD + dcol] = f2bf(oacc[t].f[r]);
        }
    }
}

// ---------------------------------------------------------------------------
// Kernel 3: out = ctx @ Wo^T + bo.  grid = (64, 8), block = 256.
// ---------------------------------------------------------------------------
__global__ __launch_bounds__(256) void mha_outproj_kernel(
    const unsigned short* __restrict__ ctx, const unsigned short* __restrict__ Wo,
    const float* __restrict__ bo, float* __restrict__ out) {
    const int lane = threadIdx.x & 31;
    const int wave = threadIdx.x >> 5;
    const int m0   = (blockIdx.x * 8 + wave) * 16;
    const int n0   = blockIdx.y * 64;

    FragC acc[4];
#pragma unroll
    for (int t = 0; t < 4; ++t)
#pragma unroll
        for (int i = 0; i < 8; ++i) acc[t].f[i] = 0.0f;

    const int rowA = m0 + (lane & 15);
    for (int k0 = 0; k0 < kD; k0 += 32) {
        v16bf a = load_frag_bf16(ctx, kD, rowA, k0, lane);
#pragma unroll
        for (int t = 0; t < 4; ++t) {
            v16bf bf = load_frag_bf16(Wo, kD, n0 + t * 16 + (lane & 15), k0, lane);
            acc[t].v = wmma_bf16(a, bf, acc[t].v);
        }
    }

    const int hi8 = (lane >> 4) << 3;
    const int col = lane & 15;
#pragma unroll
    for (int t = 0; t < 4; ++t) {
#pragma unroll
        for (int r = 0; r < 8; ++r) {
            int mrow = m0 + r + hi8;
            int ncol = n0 + t * 16 + col;
            out[(size_t)mrow * kD + ncol] = acc[t].f[r] + bo[ncol];
        }
    }
}

// ---------------------------------------------------------------------------
extern "C" void kernel_launch(void* const* d_in, const int* in_sizes, int n_in,
                              void* d_out, int out_size, void* d_ws,
                              size_t ws_size, hipStream_t stream) {
    const float* q    = (const float*)d_in[0];
    const float* k    = (const float*)d_in[1];
    const float* v    = (const float*)d_in[2];
    const float* mask = (const float*)d_in[3];
    const float* Wq   = (const float*)d_in[4];
    const float* bq   = (const float*)d_in[5];
    const float* Wk   = (const float*)d_in[6];
    const float* bk   = (const float*)d_in[7];
    const float* Wv   = (const float*)d_in[8];
    const float* bv   = (const float*)d_in[9];
    const float* Wo   = (const float*)d_in[10];
    const float* bo   = (const float*)d_in[11];

    const size_t nTok = (size_t)kB * kS * kD;  // 4,194,304 elements
    const size_t nW   = (size_t)kD * kD;       // 262,144 elements

    unsigned short* p   = (unsigned short*)d_ws;
    unsigned short* Qb  = p;           p += nTok;
    unsigned short* Kb  = p;           p += nTok;
    unsigned short* Vt  = p;           p += nTok;
    unsigned short* ctx = p;           p += nTok;
    unsigned short* Xq  = p;           p += nTok;
    unsigned short* Xk  = p;           p += nTok;
    unsigned short* Xv  = p;           p += nTok;
    unsigned short* Wqb = p;           p += nW;
    unsigned short* Wkb = p;           p += nW;
    unsigned short* Wvb = p;           p += nW;
    unsigned short* Wob = p;           p += nW;

    float* out  = (float*)d_out;
    float* attn = out + nTok;  // second tuple element, concatenated

    dim3 blk(256);
    // one-time f32 -> bf16 conversions (bandwidth-bound, ~28 MiB total)
    const int tok8 = (int)(nTok / 8), w8 = (int)(nW / 8);
    mha_cvt_kernel<<<dim3((tok8 + 255) / 256), blk, 0, stream>>>(q, Xq, tok8);
    mha_cvt_kernel<<<dim3((tok8 + 255) / 256), blk, 0, stream>>>(k, Xk, tok8);
    mha_cvt_kernel<<<dim3((tok8 + 255) / 256), blk, 0, stream>>>(v, Xv, tok8);
    mha_cvt_kernel<<<dim3((w8 + 255) / 256), blk, 0, stream>>>(Wq, Wqb, w8);
    mha_cvt_kernel<<<dim3((w8 + 255) / 256), blk, 0, stream>>>(Wk, Wkb, w8);
    mha_cvt_kernel<<<dim3((w8 + 255) / 256), blk, 0, stream>>>(Wv, Wvb, w8);
    mha_cvt_kernel<<<dim3((w8 + 255) / 256), blk, 0, stream>>>(Wo, Wob, w8);

    mha_proj_kernel<<<dim3(64, 8, 3), blk, 0, stream>>>(
        Xq, Xk, Xv, Wqb, Wkb, Wvb, bq, bk, bv, Qb, Kb, Vt);
    mha_attn_kernel<<<dim3(16, 32), blk, 0, stream>>>(
        Qb, Kb, Vt, mask, attn, ctx);
    mha_outproj_kernel<<<dim3(64, 8), blk, 0, stream>>>(ctx, Wob, bo, out);
}
